// PerWorldGRUBaseline_70789650973167
// MI455X (gfx1250) — matrix operand order repaired and checked by reading, not compiled, not run on hardware
//
#include <hip/hip_runtime.h>
#include <hip/hip_bf16.h>

// ---------------------------------------------------------------------------
// PerWorldGRU on MI455X (gfx1250): weight-stationary WMMA recurrence.
//   B=131072, N_PARAMS=32, HIDDEN=128, 3*H=384 gates, 41 steps, N_OBS=3
// Per workgroup (256 thr = 8 waves): 128 batch rows, all 41 steps.
//   - h in LDS as bf16 WMMA A-fragments (conflict-free ds_load_b128)
//   - W_hh bf16 B-fragments persistent in VGPRs (6 N-tiles x 4 k per wave)
//   - gate phase: per-thread column-pair ownership, all weights in registers,
//     hardware v_tanh_f32 for tanh AND sigmoid (1 trans op each),
//     y = h@Wo^T folded in via ds_add_f32 atomics into ping-pong y buffers
//   - LDS: weight-staging region aliased with gh buffer (dead after Breg
//     load) -> 143 KB/WG, allowing 2 workgroups per 320 KB WGP
// ---------------------------------------------------------------------------

typedef __attribute__((ext_vector_type(16))) __bf16 v16bf;
typedef __attribute__((ext_vector_type(2)))  __bf16 v2bf;
typedef __attribute__((ext_vector_type(8)))  float  v8f;

#define HIDDEN 128
#define NPAR   32
#define NOBS   3
#define STEPS  41
#define ROWS_WG 128
#define GPITCH  388   // gh pitch in bf16 elements (row*GPITCH stays dword-even)

// LDS layout (bytes).  Region 0 is a union:
//   setup phase : W-fragment staging (w1B/w2B, then whhB)  98304 B
//   main loop   : ghL bf16 gate buffer                     99328 B
constexpr int SZ_GW  = ROWS_WG * GPITCH * 2;     // 99328 (>= 98304 staging)
constexpr int SZ_HA  = 8 * 4 * 32 * 8 * 4;       // h A-fragments, 32768
constexpr int OFF_GW = 0;
constexpr int OFF_HA = OFF_GW + SZ_GW;           // 99328 (16B aligned)
constexpr int OFF_SM = OFF_HA + SZ_HA;           // 132096
// small f32 arrays (element offsets inside the smalls region)
constexpr int SM_B1   = 0;            // 128
constexpr int SM_B2   = 128;          // 128
constexpr int SM_BIH  = 256;          // 384
constexpr int SM_BHH  = 640;          // 384
constexpr int SM_WIH  = 1024;         // 384*3 = 1152
constexpr int SM_WO   = 2176;         // 3*128 = 384
constexpr int SM_BO   = 2560;         // 4
constexpr int SM_YB0  = 2564;         // 128*4 = 512 (stride-4, 16B loads)
constexpr int SM_YB1  = 3076;         // 128*4 = 512
constexpr int N_SM    = 3588;
constexpr int SMEM_BYTES = OFF_SM + N_SM * 4;    // 146448 B -> 2 WGs/WGP

// ---- native bf16 conversion helpers (backend emits v_cvt_pk_bf16_f32) ----
__device__ __forceinline__ unsigned pack2(float a, float b) {
  v2bf p; p.x = (__bf16)a; p.y = (__bf16)b;
  return __builtin_bit_cast(unsigned, p);
}
__device__ __forceinline__ float bf_lo(unsigned u) {
  v2bf p = __builtin_bit_cast(v2bf, u); return (float)p.x;
}
__device__ __forceinline__ float bf_hi(unsigned u) {
  v2bf p = __builtin_bit_cast(v2bf, u); return (float)p.y;
}

// ---- gfx1250 hardware v_tanh_f32: 1 trans op per activation ----
__device__ __forceinline__ float tanh_hw(float x) {
#if __has_builtin(__builtin_amdgcn_tanhf)
  return __builtin_amdgcn_tanhf(x);
#else
  return 2.f / (1.f + __expf(-2.f * x)) - 1.f;
#endif
}
__device__ __forceinline__ float sigmf(float x) { return 0.5f + 0.5f * tanh_hw(0.5f * x); }

// A-fragment dword index: [mtile][kstep][lane][dw]
__device__ __forceinline__ int haDW(int mt, int ks, int lane, int dw) {
  return ((mt * 4 + ks) * 32 + lane) * 8 + dw;
}
// A-fragment dword slot holding bf16 pair (col, col+1), col even.
// A 16x32 layout used by this code (contiguous k, matches B): lane = (row&15) + half*16 ; element i -> k = i + half*16
__device__ __forceinline__ int haSlot(int row, int col) {
  int mt = row >> 4, ks = col >> 5, kk = col & 31;
  int half = (kk >> 4) & 1;
  int i = kk & 15;
  return ((mt * 4 + ks) * 32 + (row & 15) + half * 16) * 8 + (i >> 1);
}

// GRU unit with all weights/biases in registers. wi = {Wr0..2, Wz0..2, Wn0..2}
__device__ __forceinline__ float gru_unit(float cr, float cz, float bin, float bhn,
                                          const float* wi,
                                          float y0, float y1, float y2,
                                          float grv, float gzv, float gnv, float hold) {
  float r = sigmf(cr + y0 * wi[0] + y1 * wi[1] + y2 * wi[2] + grv);
  float z = sigmf(cz + y0 * wi[3] + y1 * wi[4] + y2 * wi[5] + gzv);
  float n = tanh_hw(bin + y0 * wi[6] + y1 * wi[7] + y2 * wi[8] + r * (gnv + bhn));
  return (1.f - z) * n + z * hold;
}

union AFrag { unsigned u[8]; v16bf v; };

__global__ __launch_bounds__(256, 1)
void gru_wmma_persistent(const float* __restrict__ theta,
                         const float* __restrict__ W1,  const float* __restrict__ b1,
                         const float* __restrict__ W2,  const float* __restrict__ b2,
                         const float* __restrict__ Wih, const float* __restrict__ Whh,
                         const float* __restrict__ bih, const float* __restrict__ bhh,
                         const float* __restrict__ Wo,  const float* __restrict__ bo,
                         float* __restrict__ out) {
  extern __shared__ char smem[];
  unsigned* whhB = (unsigned*)(smem + OFF_GW);   // staging view (setup only)
  __bf16*   ghL  = (__bf16*)(smem + OFF_GW);     // gate-buffer view (main loop)
  unsigned* hA   = (unsigned*)(smem + OFF_HA);
  __bf16*   hA16 = (__bf16*)(smem + OFF_HA);
  float*    sm   = (float*)(smem + OFF_SM);

  const int tid  = threadIdx.x;
  const int lane = tid & 31;
  const int w    = tid >> 5;
  const int mg   = w & 1;   // M-group: 0..1 (4 mtiles each)
  const int ng   = w >> 1;  // N-group: 0..3 (6 ntiles each)
  const int wgRow0 = blockIdx.x * ROWS_WG;

  // ---- stage small params + zero y0 buffer ----
  for (int i = tid; i < 128; i += 256) { sm[SM_B1 + i] = b1[i]; sm[SM_B2 + i] = b2[i]; }
  for (int i = tid; i < 384; i += 256) { sm[SM_BIH + i] = bih[i]; sm[SM_BHH + i] = bhh[i]; }
  for (int i = tid; i < 384 * 3; i += 256) sm[SM_WIH + i] = Wih[i];
  for (int i = tid; i < 3 * 128; i += 256) sm[SM_WO + i] = Wo[i];
  if (tid < 3) sm[SM_BO + tid] = bo[tid];
  sm[SM_YB0 + tid] = 0.f;  sm[SM_YB0 + 256 + tid] = 0.f;

  // ---- build W1/W2 bf16 B-fragments (staging region) ----
  // B layout used by this code (KxN, bf16, contiguous k, same as A): lane = n%16 + half*16 ; element i -> k = i + half*16
  unsigned* w1B = whhB;           // 8 tiles * 1 k  * 32 * 8 dw
  unsigned* w2B = whhB + 2048;    // 8 tiles * 4 k  * 32 * 8 dw
  for (int idx = tid; idx < 8 * 32 * 8; idx += 256) {
    int dw = idx & 7, ln = (idx >> 3) & 31, nt = idx >> 8;
    int n = nt * 16 + (ln & 15);
    int k0 = (ln >> 4) * 16 + dw * 2;
    w1B[idx] = pack2(W1[n * NPAR + k0], W1[n * NPAR + k0 + 1]);
  }
  for (int idx = tid; idx < 8 * 4 * 32 * 8; idx += 256) {
    int dw = idx & 7, ln = (idx >> 3) & 31, ks = (idx >> 8) & 3, nt = idx >> 10;
    int n = nt * 16 + (ln & 15);
    int k0 = ks * 32 + (ln >> 4) * 16 + dw * 2;
    w2B[idx] = pack2(W2[n * HIDDEN + k0], W2[n * HIDDEN + k0 + 1]);
  }
  __syncthreads();

  // ---- encoder GEMM1: t1 = relu(theta @ W1^T + b1) -> hA (A-fragment bf16) ----
  const int half = lane >> 4;
  for (int m = 0; m < 4; ++m) {
    int mt = mg * 4 + m;
    AFrag af;
    int row_g = wgRow0 + mt * 16 + (lane & 15);
#pragma unroll
    for (int dw = 0; dw < 8; ++dw) {
      int k0 = 2 * dw + half * 16;     // A-layout k for pair
      af.u[dw] = pack2(theta[row_g * NPAR + k0], theta[row_g * NPAR + k0 + 1]);
    }
#pragma unroll
    for (int t = 0; t < 2; ++t) {
      int nt = ng * 2 + t;
      v16bf bf = *(const v16bf*)(w1B + (nt * 32 + lane) * 8);
      v8f acc = {0.f, 0.f, 0.f, 0.f, 0.f, 0.f, 0.f, 0.f};
      acc = __builtin_amdgcn_wmma_f32_16x16x32_bf16(false, af.v, false, bf, (short)0, acc, false, false);
      int col  = nt * 16 + (lane & 15);
      float bb = sm[SM_B1 + col];
      int rowb = mt * 16 + half * 8;
#pragma unroll
      for (int v = 0; v < 8; ++v) {
        float x = fmaxf(acc[v] + bb, 0.f);
        hA16[haSlot(rowb + v, col & ~1) * 2 + (col & 1)] = (__bf16)x;
      }
    }
  }
  __syncthreads();

  // ---- encoder GEMM2: h0 = t1 @ W2^T + b2 (accumulate in regs, then overwrite hA) ----
  v8f accH[4][2];
  for (int m = 0; m < 4; ++m) {
    int mt = mg * 4 + m;
    v16bf a2[4];
#pragma unroll
    for (int ks = 0; ks < 4; ++ks) a2[ks] = *(const v16bf*)(hA + haDW(mt, ks, lane, 0));
#pragma unroll
    for (int t = 0; t < 2; ++t) {
      int nt = ng * 2 + t;
      v8f acc = {0.f, 0.f, 0.f, 0.f, 0.f, 0.f, 0.f, 0.f};
#pragma unroll
      for (int ks = 0; ks < 4; ++ks) {
        v16bf bf = *(const v16bf*)(w2B + ((nt * 4 + ks) * 32 + lane) * 8);
        acc = __builtin_amdgcn_wmma_f32_16x16x32_bf16(false, a2[ks], false, bf, (short)0, acc, false, false);
      }
      accH[m][t] = acc;
    }
  }
  __syncthreads();   // all GEMM2 reads of hA / w2B done
  for (int m = 0; m < 4; ++m) {
    int mt = mg * 4 + m;
#pragma unroll
    for (int t = 0; t < 2; ++t) {
      int nt = ng * 2 + t;
      int col  = nt * 16 + (lane & 15);
      float bb = sm[SM_B2 + col];
      int rowb = mt * 16 + half * 8;
#pragma unroll
      for (int v = 0; v < 8; ++v)
        hA16[haSlot(rowb + v, col & ~1) * 2 + (col & 1)] = (__bf16)(accH[m][t][v] + bb);
    }
  }
  // ---- build W_hh B-fragments (staging region, overwrites w1B/w2B) ----
  for (int idx = tid; idx < 24 * 4 * 32 * 8; idx += 256) {
    int dw = idx & 7, ln = (idx >> 3) & 31, ks = (idx >> 8) & 3, nt = idx >> 10;
    int n = nt * 16 + (ln & 15);
    int k0 = ks * 32 + (ln >> 4) * 16 + dw * 2;
    whhB[idx] = pack2(Whh[n * HIDDEN + k0], Whh[n * HIDDEN + k0 + 1]);
  }
  __syncthreads();

  // ---- W_hh fragments persistent in registers: 6 N-tiles x 4 k = 192 VGPRs ----
  // After this point the staging region is dead and is reused as ghL.
  v16bf Breg[6][4];
#pragma unroll
  for (int t = 0; t < 6; ++t)
#pragma unroll
    for (int ks = 0; ks < 4; ++ks)
      Breg[t][ks] = *(const v16bf*)(whhB + (((ng * 6 + t) * 4 + ks) * 32 + lane) * 8);

  // ---- per-thread gate constants: this thread owns columns (jcol, jcol+1)
  //      for rows rowb0 + 4*it, for ALL steps. Everything below is invariant.
  const int jcol  = (tid & 63) * 2;
  const int rowb0 = tid >> 6;
  float C0[4], C1[4], WI0[9], WI1[9], WOv0[3], WOv1[3];
#pragma unroll
  for (int s = 0; s < 2; ++s) {
    int jj = jcol + s;
    float* C  = s ? C1  : C0;
    float* WI = s ? WI1 : WI0;
    float* WO = s ? WOv1 : WOv0;
    C[0] = sm[SM_BIH + jj]       + sm[SM_BHH + jj];        // r: bih+bhh folded
    C[1] = sm[SM_BIH + 128 + jj] + sm[SM_BHH + 128 + jj];  // z: bih+bhh folded
    C[2] = sm[SM_BIH + 256 + jj];                          // n: bih
    C[3] = sm[SM_BHH + 256 + jj];                          // n: bhh (inside r*)
#pragma unroll
    for (int g = 0; g < 3; ++g)
#pragma unroll
      for (int o = 0; o < 3; ++o)
        WI[g * 3 + o] = sm[SM_WIH + (g * 128 + jj) * 3 + o];
#pragma unroll
    for (int o = 0; o < 3; ++o)
      WO[o] = sm[SM_WO + o * 128 + jj];
  }
  __syncthreads();

  // ================= recurrence: 41 steps =================
  for (int step = 0; step < STEPS; ++step) {
    const int ybPrev = (step & 1) ? SM_YB1 : SM_YB0;
    const int ybNew  = (step & 1) ? SM_YB0 : SM_YB1;
    // zero this step's y accumulator (raced only by this phase; barrier below)
    sm[ybNew + tid] = 0.f;  sm[ybNew + 256 + tid] = 0.f;

    // Phase A: gh = h @ W_hh^T   (96 WMMAs/wave, weights stationary)
    for (int m = 0; m < 4; ++m) {
      int mt = mg * 4 + m;
      v16bf a[4];
#pragma unroll
      for (int ks = 0; ks < 4; ++ks) a[ks] = *(const v16bf*)(hA + haDW(mt, ks, lane, 0));
#pragma unroll
      for (int t = 0; t < 6; ++t) {
        int nt = ng * 6 + t;
        v8f acc = {0.f, 0.f, 0.f, 0.f, 0.f, 0.f, 0.f, 0.f};
#pragma unroll
        for (int ks = 0; ks < 4; ++ks)
          acc = __builtin_amdgcn_wmma_f32_16x16x32_bf16(false, a[ks], false, Breg[t][ks], (short)0, acc, false, false);
        int col  = nt * 16 + (lane & 15);
        int rowb = mt * 16 + half * 8;
#pragma unroll
        for (int v = 0; v < 8; v += 2) {   // one packed cvt per row-pair
          v2bf pp = __builtin_bit_cast(v2bf, pack2(acc[v], acc[v + 1]));
          ghL[(rowb + v)     * GPITCH + col] = pp.x;
          ghL[(rowb + v + 1) * GPITCH + col] = pp.y;
        }
      }
    }
    __syncthreads();

    // Phase B: gates + y-contribution via LDS float atomics
    for (int it = 0; it < 32; ++it) {
      int row = rowb0 + it * 4;
      int gb  = row * GPITCH;
      unsigned gr = *(const unsigned*)(ghL + gb + jcol);
      unsigned gz = *(const unsigned*)(ghL + gb + 128 + jcol);
      unsigned gn = *(const unsigned*)(ghL + gb + 256 + jcol);
      int slot = haSlot(row, jcol);
      unsigned hu = hA[slot];
      float4 yv = *(const float4*)(sm + ybPrev + row * 4);
      float hn0 = gru_unit(C0[0], C0[1], C0[2], C0[3], WI0, yv.x, yv.y, yv.z,
                           bf_lo(gr), bf_lo(gz), bf_lo(gn), bf_lo(hu));
      float hn1 = gru_unit(C1[0], C1[1], C1[2], C1[3], WI1, yv.x, yv.y, yv.z,
                           bf_hi(gr), bf_hi(gz), bf_hi(gn), bf_hi(hu));
      hA[slot] = pack2(hn0, hn1);
      atomicAdd(sm + ybNew + row * 4 + 0, hn0 * WOv0[0] + hn1 * WOv1[0]);
      atomicAdd(sm + ybNew + row * 4 + 1, hn0 * WOv0[1] + hn1 * WOv1[1]);
      atomicAdd(sm + ybNew + row * 4 + 2, hn0 * WOv0[2] + hn1 * WOv1[2]);
    }
    __syncthreads();

    // Final: add bias, publish y (used as next step's input) + store to HBM
    if (tid < 128) {
      int row = tid;
      float yo0 = sm[ybNew + row * 4 + 0] + sm[SM_BO + 0];
      float yo1 = sm[ybNew + row * 4 + 1] + sm[SM_BO + 1];
      float yo2 = sm[ybNew + row * 4 + 2] + sm[SM_BO + 2];
      sm[ybNew + row * 4 + 0] = yo0;
      sm[ybNew + row * 4 + 1] = yo1;
      sm[ybNew + row * 4 + 2] = yo2;
      size_t gbase = (size_t)(wgRow0 + row) * (STEPS * NOBS) + (size_t)step * NOBS;
      out[gbase + 0] = yo0; out[gbase + 1] = yo1; out[gbase + 2] = yo2;
    }
    __syncthreads();
  }
}

extern "C" void kernel_launch(void* const* d_in, const int* in_sizes, int n_in,
                              void* d_out, int out_size, void* d_ws, size_t ws_size,
                              hipStream_t stream) {
  const float* theta = (const float*)d_in[0];
  const float* W1    = (const float*)d_in[1];
  const float* b1    = (const float*)d_in[2];
  const float* W2    = (const float*)d_in[3];
  const float* b2    = (const float*)d_in[4];
  const float* Wih   = (const float*)d_in[5];
  const float* Whh   = (const float*)d_in[6];
  const float* bih   = (const float*)d_in[7];
  const float* bhh   = (const float*)d_in[8];
  const float* Wo    = (const float*)d_in[9];
  const float* bo    = (const float*)d_in[10];
  float* out = (float*)d_out;

  int batch = in_sizes[0] / NPAR;
  int grid = batch / ROWS_WG;
  if (grid < 1) grid = 1;

  (void)hipFuncSetAttribute((const void*)gru_wmma_persistent,
                            hipFuncAttributeMaxDynamicSharedMemorySize, SMEM_BYTES);
  gru_wmma_persistent<<<grid, 256, SMEM_BYTES, stream>>>(
      theta, W1, b1, W2, b2, Wih, Whh, bih, bhh, Wo, bo, out);
}
